// GNNEncoder_21148418966315
// MI455X (gfx1250) — compile-verified, matrix operand order checked by reading, not run on hardware
//
#include <hip/hip_runtime.h>
#include <hip/hip_bf16.h>

typedef __bf16 bf16t;
typedef __attribute__((ext_vector_type(16))) __bf16 v16bf;
typedef __attribute__((ext_vector_type(8)))  float  v8f;
typedef __attribute__((ext_vector_type(4)))  unsigned int v4u;   // native 16B vector

// ---------------------------------------------------------------------------
// utility kernels
// ---------------------------------------------------------------------------
__global__ void k_zero(float* __restrict__ p, size_t n) {
  size_t i = (size_t)blockIdx.x * 256 + threadIdx.x;
  if (i < n) p[i] = 0.0f;
}

// W: [K, D] row-major f32  ->  Wt: [D, K] row-major bf16
__global__ void k_wt(const float* __restrict__ W, bf16t* __restrict__ Wt, int K, int D) {
  int i = blockIdx.x * 256 + threadIdx.x;
  if (i < K * D) {
    int c = i / K, k = i % K;
    Wt[i] = (bf16t)W[(size_t)k * D + c];
  }
}

// per-graph node counts for mean pool
__global__ void k_count(const long long* __restrict__ batch, float* __restrict__ cnt, int N) {
  int i = blockIdx.x * 256 + threadIdx.x;
  if (i < N) atomicAdd(&cnt[(size_t)batch[i]], 1.0f);
}

__global__ void k_div(float* __restrict__ out, const float* __restrict__ cnt, int G) {
  int i = blockIdx.x * 256 + threadIdx.x;
  if (i < G * 256) {
    float c = cnt[i >> 8];
    out[i] = out[i] / (c > 1.0f ? c : 1.0f);
  }
}

// ---------------------------------------------------------------------------
// edge message + scatter-add:  aggr[dst] += relu(x[src] + edge_attr @ We + be)
// one wave per edge, 4 edges per wave, We/be staged in LDS
// ---------------------------------------------------------------------------
template <int D>
__global__ __launch_bounds__(256) void k_edge(
    const float* __restrict__ x, const float* __restrict__ ea,
    const long long* __restrict__ src, const long long* __restrict__ dst,
    const float* __restrict__ We, const float* __restrict__ be,
    float* __restrict__ aggr, int E)
{
  __shared__ float sW[16 * D];
  __shared__ float sBias[D];
  for (int i = threadIdx.x; i < 16 * D; i += 256) sW[i] = We[i];
  for (int i = threadIdx.x; i < D; i += 256) sBias[i] = be[i];
  __syncthreads();

  const int wave = threadIdx.x >> 5;
  const int lane = threadIdx.x & 31;

  for (int i = 0; i < 4; ++i) {
    int e = (blockIdx.x * 8 + wave) * 4 + i;          // uniform per wave
    if (e >= E) continue;
    long long s = src[e];
    long long d = dst[e];
    float eav = (lane < 16) ? ea[(size_t)e * 16 + lane] : 0.0f;
#pragma unroll
    for (int j = 0; j < D / 32; ++j) {
      int f = lane + j * 32;
      float acc = sBias[f];
#pragma unroll
      for (int k = 0; k < 16; ++k)
        acc += __shfl(eav, k, 32) * sW[k * D + f];
      float m = x[(size_t)s * D + f] + acc;
      m = m > 0.0f ? m : 0.0f;
      atomicAdd(&aggr[(size_t)d * D + f], m);
    }
  }
}

// ---------------------------------------------------------------------------
// bf16 WMMA GEMM, block tile 128 x 256 (full output width), BK = 32.
// A: [N,K] f32 (converted to bf16 on the fly) or bf16.  Bt: [256,K] bf16.
// Epilogue: +bias, relu, then any of:
//   outB       : bf16 store (feeds next GEMM)
//   outF/outF2 : f32 store(s) (feeds edge layer / residual accumulator)
//   poolSums   : fused mean-pool atomic accumulate keyed by batch[row]
// 8 waves; wave (wm,wn) owns 64 rows x 64 cols -> 4x4 wmma accumulators.
// ---------------------------------------------------------------------------
template <typename AT>
__global__ __launch_bounds__(256) void k_gemm(
    const AT* __restrict__ A, const bf16t* __restrict__ Bt,
    const float* __restrict__ bias, int N, int K,
    bf16t* __restrict__ outB, float* __restrict__ outF, float* __restrict__ outF2,
    float* __restrict__ poolSums, const long long* __restrict__ batch)
{
  __shared__ bf16t sA[128][40];   // stride 40 -> 80B rows, 16B-aligned frag reads
  __shared__ bf16t sB[256][40];

  const int tid  = threadIdx.x;
  const int wave = tid >> 5;
  const int lane = tid & 31;
  const int lm   = lane & 15;
  const int lh   = lane >> 4;
  const int wm   = wave & 1;      // 0..1 : 64-row half
  const int wn   = wave >> 1;     // 0..3 : 64-col quarter
  const int rowBase = blockIdx.x * 128;

  v8f zero = {};
  v8f acc[4][4];
#pragma unroll
  for (int mt = 0; mt < 4; ++mt)
#pragma unroll
    for (int nt = 0; nt < 4; ++nt) acc[mt][nt] = zero;

  const int ra = tid >> 1;       // 0..127 : A-tile row this thread fills
  const int hf = tid & 1;        // 16-element half of the 32-wide K chunk
  int gr = rowBase + ra; if (gr >= N) gr = N - 1;   // clamp; stores guarded later
  const int rb = tid;            // 0..255 : B-tile row (output column)

  for (int k0 = 0; k0 < K; k0 += 32) {
    // ---- fetch A chunk (16 elements) into registers, as bf16 ----
    union { v4u q[2]; bf16t h[16]; } ua;
    if constexpr (sizeof(AT) == 4) {
      const v4u* pa = (const v4u*)((const float*)A + (size_t)gr * K + k0 + hf * 16);
      union { v4u q[4]; float f[16]; } uf;
      uf.q[0] = __builtin_nontemporal_load(pa + 0);
      uf.q[1] = __builtin_nontemporal_load(pa + 1);
      uf.q[2] = __builtin_nontemporal_load(pa + 2);
      uf.q[3] = __builtin_nontemporal_load(pa + 3);
#pragma unroll
      for (int j = 0; j < 16; ++j) ua.h[j] = (bf16t)uf.f[j];
    } else {
      const v4u* pa = (const v4u*)((const bf16t*)A + (size_t)gr * K + k0 + hf * 16);
      ua.q[0] = __builtin_nontemporal_load(pa + 0);
      ua.q[1] = __builtin_nontemporal_load(pa + 1);
    }
    // ---- fetch B chunk (32 bf16 = one output column's K-slice) ----
    const v4u* pb = (const v4u*)(Bt + (size_t)rb * K + k0);
    v4u b0 = pb[0], b1 = pb[1], b2 = pb[2], b3 = pb[3];

    __syncthreads();
    *(v4u*)&sA[ra][hf * 16]     = ua.q[0];
    *(v4u*)&sA[ra][hf * 16 + 8] = ua.q[1];
    *(v4u*)&sB[rb][0]  = b0;
    *(v4u*)&sB[rb][8]  = b1;
    *(v4u*)&sB[rb][16] = b2;
    *(v4u*)&sB[rb][24] = b3;
    __syncthreads();

    // ---- fragments: A lane L holds M=L%16, K in {k0..k0+7, 16+k0..+7}, k0=(L/16)*8
    v16bf fa[4], fb[4];
#pragma unroll
    for (int mt = 0; mt < 4; ++mt) {
      int rr = wm * 64 + mt * 16 + lm;
      union { v4u q[2]; v16bf v; } u;
      u.q[0] = *(const v4u*)&sA[rr][lh * 8];
      u.q[1] = *(const v4u*)&sA[rr][16 + lh * 8];
      fa[mt] = u.v;
    }
#pragma unroll
    for (int nt = 0; nt < 4; ++nt) {
      int cc = wn * 64 + nt * 16 + lm;
      union { v4u q[2]; v16bf v; } u;
      u.q[0] = *(const v4u*)&sB[cc][lh * 8];
      u.q[1] = *(const v4u*)&sB[cc][16 + lh * 8];
      fb[nt] = u.v;
    }
#pragma unroll
    for (int mt = 0; mt < 4; ++mt)
#pragma unroll
      for (int nt = 0; nt < 4; ++nt)
        acc[mt][nt] = __builtin_amdgcn_wmma_f32_16x16x32_bf16(
            false, fa[mt], false, fb[nt], (short)0, acc[mt][nt], false, false);
  }

  // ---- epilogue: bias + relu + store/pool ----
  float bv[4];
#pragma unroll
  for (int nt = 0; nt < 4; ++nt) bv[nt] = bias[wn * 64 + nt * 16 + lm];

#pragma unroll
  for (int mt = 0; mt < 4; ++mt) {
#pragma unroll
    for (int e = 0; e < 8; ++e) {
      int row = rowBase + wm * 64 + mt * 16 + lh * 8 + e;
      if (row < N) {
        long long g = poolSums ? batch[row] : 0;
#pragma unroll
        for (int nt = 0; nt < 4; ++nt) {
          int col = wn * 64 + nt * 16 + lm;
          float v = acc[mt][nt][e] + bv[nt];
          v = v > 0.0f ? v : 0.0f;                 // relu after every GEMM here
          size_t off = (size_t)row * 256 + col;
          if (outB) outB[off] = (bf16t)v;
          if (outF) outF[off] = v;
          if (outF2) outF2[off] = v;
          if (poolSums) atomicAdd(&poolSums[(size_t)g * 256 + col], v);
        }
      }
    }
  }
}

// ---------------------------------------------------------------------------
extern "C" void kernel_launch(void* const* d_in, const int* in_sizes, int n_in,
                              void* d_out, int out_size, void* d_ws, size_t ws_size,
                              hipStream_t stream)
{
  const float* x    = (const float*)d_in[0];
  const float* ea   = (const float*)d_in[1];
  const float* We1  = (const float*)d_in[2];
  const float* be1  = (const float*)d_in[3];
  const float* W1a  = (const float*)d_in[4];
  const float* b1a  = (const float*)d_in[5];
  const float* W1b  = (const float*)d_in[6];
  const float* b1b  = (const float*)d_in[7];
  const float* We2  = (const float*)d_in[8];
  const float* be2  = (const float*)d_in[9];
  const float* W2a  = (const float*)d_in[10];
  const float* b2a  = (const float*)d_in[11];
  const float* W2b  = (const float*)d_in[12];
  const float* b2b  = (const float*)d_in[13];
  const long long* eidx  = (const long long*)d_in[14];
  const long long* batch = (const long long*)d_in[15];

  const int N = in_sizes[0] / 128;
  const int E = in_sizes[1] / 16;
  const int G = out_size / 256;
  const long long* src = eidx;
  const long long* dst = eidx + E;

  char* ws = (char*)d_ws;
  size_t off = 0;
  auto alloc = [&](size_t bytes) -> void* {
    void* p = ws + off;
    off += (bytes + 255) & ~(size_t)255;
    return p;
  };

  float* hpre  = (float*)alloc((size_t)N * 256 * sizeof(float)); // residual + aggr accumulator
  float* h1    = (float*)alloc((size_t)N * 256 * sizeof(float)); // layer-1 output (gather source)
  bf16t* A2buf = (bf16t*)alloc((size_t)N * 256 * sizeof(bf16t)); // bf16 inter-GEMM activations
  bf16t* Wt1a  = (bf16t*)alloc((size_t)256 * 128 * sizeof(bf16t));
  bf16t* Wt1b  = (bf16t*)alloc((size_t)256 * 256 * sizeof(bf16t));
  bf16t* Wt2a  = (bf16t*)alloc((size_t)256 * 256 * sizeof(bf16t));
  bf16t* Wt2b  = (bf16t*)alloc((size_t)256 * 256 * sizeof(bf16t));
  float* cnt   = (float*)alloc((size_t)G * sizeof(float));
  (void)ws_size; (void)n_in;

  float* sums = (float*)d_out;

  // weight transpose+convert to bf16 [D,K]
  k_wt<<<(256 * 128 + 255) / 256, 256, 0, stream>>>(W1a, Wt1a, 128, 256);
  k_wt<<<(256 * 256 + 255) / 256, 256, 0, stream>>>(W1b, Wt1b, 256, 256);
  k_wt<<<(256 * 256 + 255) / 256, 256, 0, stream>>>(W2a, Wt2a, 256, 256);
  k_wt<<<(256 * 256 + 255) / 256, 256, 0, stream>>>(W2b, Wt2b, 256, 256);

  // zero pooling buffers (consumed by fused GEMM4 epilogue)
  k_zero<<<(G * 256 + 255) / 256, 256, 0, stream>>>(sums, (size_t)G * 256);
  k_zero<<<(G + 255) / 256, 256, 0, stream>>>(cnt, (size_t)G);
  k_count<<<(N + 255) / 256, 256, 0, stream>>>(batch, cnt, N);

  dim3 gGemm((N + 127) / 128, 1);

  // ------------------- layer 1 -------------------
  (void)hipMemcpyAsync(hpre, x, (size_t)N * 128 * sizeof(float),
                       hipMemcpyDeviceToDevice, stream);
  k_edge<128><<<(E + 31) / 32, 256, 0, stream>>>(x, ea, src, dst, We1, be1, hpre, E);
  // GEMM1: f32 in [N,128] -> bf16 out
  k_gemm<float><<<gGemm, 256, 0, stream>>>(hpre, Wt1a, b1a, N, 128,
                                           A2buf, nullptr, nullptr, nullptr, nullptr);
  // GEMM2: bf16 in [N,256] -> f32 dual store (h1 + residual accumulator)
  k_gemm<bf16t><<<gGemm, 256, 0, stream>>>(A2buf, Wt1b, b1b, N, 256,
                                           nullptr, h1, hpre, nullptr, nullptr);

  // ------------------- layer 2 -------------------
  k_edge<256><<<(E + 31) / 32, 256, 0, stream>>>(h1, ea, src, dst, We2, be2, hpre, E);
  // GEMM3: f32 in [N,256] -> bf16 out
  k_gemm<float><<<gGemm, 256, 0, stream>>>(hpre, Wt2a, b2a, N, 256,
                                           A2buf, nullptr, nullptr, nullptr, nullptr);
  // GEMM4: bf16 in [N,256] -> fused mean-pool accumulate into d_out
  k_gemm<bf16t><<<gGemm, 256, 0, stream>>>(A2buf, Wt2b, b2b, N, 256,
                                           nullptr, nullptr, nullptr, sums, batch);

  // ------------------- mean pool divide -------------------
  k_div<<<(G * 256 + 255) / 256, 256, 0, stream>>>(sums, cnt, G);
}
